// OpenProtTransformerBlock_80779744903733
// MI455X (gfx1250) — matrix operand order repaired and checked by reading, not compiled, not run on hardware
//
#include <hip/hip_runtime.h>
#include <hip/hip_bf16.h>

typedef __attribute__((ext_vector_type(16))) __bf16 v16bf;
typedef __attribute__((ext_vector_type(2)))  __bf16 v2bf;
typedef __attribute__((ext_vector_type(8)))  float  v8f;

#define ASTRIDE 40

// Native fp32->bf16 convert (RNE); lowers to hardware cvt instead of
// the 3-4 VALU-op integer emulation.
__device__ __forceinline__ unsigned short f2bf(float f) {
  union { __bf16 h; unsigned short u; } c;
  c.h = (__bf16)f;
  return c.u;
}
__device__ __forceinline__ unsigned int pk2(float a, float b) {
  union { v2bf h; unsigned int u; } c;
  c.h[0] = (__bf16)a; c.h[1] = (__bf16)b;
  return c.u;
}

// A fragment (16x32 bf16): lanes 0-15 -> M=lane, K={0..7,16..23}; lanes 16-31 -> K={8..15,24..31}
__device__ __forceinline__ v16bf frag_ld_a(const unsigned short* base, int stride, int lane) {
  int mr = lane & 15, kb = (lane >> 4) << 3;
  const unsigned short* p = base + mr * stride + kb;
  union { uint4 q[2]; v16bf v; } u;
  u.q[0] = *(const uint4*)(p);
  u.q[1] = *(const uint4*)(p + 16);
  return u.v;
}
// B fragment (32x16 bf16), LDS holds B transposed: row n, 32 contiguous k.
// lanes 0-15 -> N=lane, K={0..7,16..23}; lanes 16-31 -> K={8..15,24..31}
__device__ __forceinline__ v16bf frag_ld_b(const unsigned short* base, int stride, int lane) {
  int nr = lane & 15, kb = (lane >> 4) << 3;
  const unsigned short* p = base + nr * stride + kb;
  union { uint4 q[2]; v16bf v; } u;
  u.q[0] = *(const uint4*)(p);
  u.q[1] = *(const uint4*)(p + 16);
  return u.v;
}

__device__ __forceinline__ v8f wmma_bf16(v16bf a, v16bf b, v8f c) {
  return __builtin_amdgcn_wmma_f32_16x16x32_bf16(false, a, false, b, (short)0, c, false, false);
}

// ---------------------------------------------------------------------------
// Generic GEMM: C[M,N] = epi(A[M,K] @ W[K,N] + bias)
// EPI: 0=bias 1=relu 2=sigmoid 3=sigmoid*mul*mask 4=(x)*mul+res 5=x+res
// TRANSC: store C channel-major: C[n*M + m] (contiguous float4 stores).
// mul/res always indexed row-major with ldc.
// Requires M%64==0, K%32==0. N arbitrary (zero-padded tiles, guarded stores).
// ---------------------------------------------------------------------------
template<int EPI, bool TRANSC>
__global__ void __launch_bounds__(256)
k_gemm(const float* __restrict__ A, int lda,
       const float* __restrict__ W, int ldw,
       const float* __restrict__ bias,
       float* __restrict__ C, int ldc,
       int M, int N, int K,
       const float* __restrict__ mul,
       const float* __restrict__ res,
       const float* __restrict__ maskv, int maskN)
{
  __shared__ __align__(16) unsigned short sA[64 * ASTRIDE];
  __shared__ __align__(16) unsigned short sB[64 * ASTRIDE];
  const int tid = threadIdx.x, lane = tid & 31, wave = tid >> 5;
  const int bm = blockIdx.y * 64, bn = blockIdx.x * 64;
  const int wm = wave & 3;
  const int wn2 = (wave >> 2) * 2;
  const int ra = tid >> 2, kca = (tid & 3) * 8;   // A staging coords
  const int kb = tid >> 3, nbb = (tid & 7) * 8;   // B staging coords
  v8f acc0 = {}; v8f acc1 = {};
  for (int k0 = 0; k0 < K; k0 += 32) {
    { // stage A tile 64x32 (one ds_store_b128 per thread)
      const float* gp = A + (size_t)(bm + ra) * lda + k0 + kca;
      float4 v0 = *(const float4*)gp;
      float4 v1 = *(const float4*)(gp + 4);
      uint4 pk;
      pk.x = pk2(v0.x, v0.y); pk.y = pk2(v0.z, v0.w);
      pk.z = pk2(v1.x, v1.y); pk.w = pk2(v1.z, v1.w);
      *(uint4*)&sA[ra * ASTRIDE + kca] = pk;
    }
    { // stage B tile 32x64 transposed into LDS
      const float* gp = W + (size_t)(k0 + kb) * ldw + bn + nbb;
      if (bn + 64 <= N) {
        float4 v0 = *(const float4*)gp;
        float4 v1 = *(const float4*)(gp + 4);
        sB[(nbb + 0) * ASTRIDE + kb] = f2bf(v0.x); sB[(nbb + 1) * ASTRIDE + kb] = f2bf(v0.y);
        sB[(nbb + 2) * ASTRIDE + kb] = f2bf(v0.z); sB[(nbb + 3) * ASTRIDE + kb] = f2bf(v0.w);
        sB[(nbb + 4) * ASTRIDE + kb] = f2bf(v1.x); sB[(nbb + 5) * ASTRIDE + kb] = f2bf(v1.y);
        sB[(nbb + 6) * ASTRIDE + kb] = f2bf(v1.z); sB[(nbb + 7) * ASTRIDE + kb] = f2bf(v1.w);
      } else {
#pragma unroll
        for (int e = 0; e < 8; ++e) {
          int n = bn + nbb + e;
          float v = (n < N) ? gp[e] : 0.f;
          sB[(nbb + e) * ASTRIDE + kb] = f2bf(v);
        }
      }
    }
    if (k0 + 32 < K) { // prefetch next tiles (global_prefetch_b8)
      __builtin_prefetch(A + (size_t)(bm + ra) * lda + k0 + 32 + kca, 0, 1);
      __builtin_prefetch(W + (size_t)(k0 + 32 + kb) * ldw + bn + nbb, 0, 1);
    }
    __syncthreads();
    v16bf af  = frag_ld_a(&sA[(wm * 16) * ASTRIDE], ASTRIDE, lane);
    v16bf bf0 = frag_ld_b(&sB[(wn2 * 16) * ASTRIDE], ASTRIDE, lane);
    v16bf bf1 = frag_ld_b(&sB[((wn2 + 1) * 16) * ASTRIDE], ASTRIDE, lane);
    acc0 = wmma_bf16(af, bf0, acc0);
    acc1 = wmma_bf16(af, bf1, acc1);
    __syncthreads();
  }
#pragma unroll
  for (int half = 0; half < 2; ++half) {
    v8f acc = half ? acc1 : acc0;
    int n = bn + (wn2 + half) * 16 + (lane & 15);
    if (n >= N) continue;
    float bv = bias ? bias[n] : 0.f;
    int mbase = bm + wm * 16 + ((lane >> 4) << 3);
    float vals[8];
#pragma unroll
    for (int r = 0; r < 8; ++r) {
      int m = mbase + r;
      size_t idxR = (size_t)m * ldc + n;   // row-major index for mul/res
      float v = acc[r] + bv;
      if (EPI == 1) v = fmaxf(v, 0.f);
      else if (EPI == 2) v = 1.f / (1.f + __expf(-v));
      else if (EPI == 3) {
        float s = 1.f / (1.f + __expf(-v));
        float mf = 1.f;
        if (maskv) { int i = m / maskN, j = m % maskN; mf = maskv[i] * maskv[j]; }
        v = s * mul[idxR] * mf;
      } else if (EPI == 4) v = v * mul[idxR] + res[idxR];
      else if (EPI == 5) v = v + res[idxR];
      vals[r] = v;
    }
    if (TRANSC) {
      float* cp = C + (size_t)n * M + mbase;
      *(float4*)cp       = make_float4(vals[0], vals[1], vals[2], vals[3]);
      *(float4*)(cp + 4) = make_float4(vals[4], vals[5], vals[6], vals[7]);
    } else {
#pragma unroll
      for (int r = 0; r < 8; ++r) C[(size_t)(mbase + r) * ldc + n] = vals[r];
    }
  }
}

// ---------------------------------------------------------------------------
// Triangle multiplication: per-channel GEMM with CHANNEL-MAJOR inputs
// At/Bt layout: T[c][r] with r = row*512 + col of the original (N,N,C) tensor.
// OUTG: X[i,j,c] = sum_k A[i,k,c]*B[j,k,c]; else X[i,j,c] = sum_k A[k,i,c]*B[k,j,c]
// grid (c=128, jt=16, it=8); block 256 (8 waves), tile 64x32.
// Output written back (i,j,c)-major; c-fastest grid order gives L2 write merge.
// ---------------------------------------------------------------------------
template<bool OUTG>
__global__ void __launch_bounds__(256)
k_tri_mm(const float* __restrict__ A, const float* __restrict__ Bm, float* __restrict__ X)
{
  __shared__ __align__(16) unsigned short sA[64 * ASTRIDE];
  __shared__ __align__(16) unsigned short sB[32 * ASTRIDE];
  const int tid = threadIdx.x, lane = tid & 31, wave = tid >> 5;
  const int c = blockIdx.x;
  const int bj = blockIdx.y * 32, bi = blockIdx.z * 64;
  const int wm = wave & 3, wn = wave >> 2;
  const size_t cbase = (size_t)c * 262144;
  v8f acc = {};
  for (int k0 = 0; k0 < 512; k0 += 32) {
    if (OUTG) {
      { // sA[r][k] = At[c][(bi+r)*512 + k] : contiguous k
        int r = tid >> 2, kc = (tid & 3) * 8;
        const float* gp = A + cbase + (size_t)(bi + r) * 512 + k0 + kc;
        float4 v0 = *(const float4*)gp;
        float4 v1 = *(const float4*)(gp + 4);
        uint4 pk;
        pk.x = pk2(v0.x, v0.y); pk.y = pk2(v0.z, v0.w);
        pk.z = pk2(v1.x, v1.y); pk.w = pk2(v1.z, v1.w);
        *(uint4*)&sA[r * ASTRIDE + kc] = pk;
        if (k0 + 32 < 512) __builtin_prefetch(gp + 32, 0, 1);
      }
      { // sB[j][k] = Bt[c][(bj+j)*512 + k] : contiguous k
        int j = tid >> 3, kc = (tid & 7) * 4;
        const float* gp = Bm + cbase + (size_t)(bj + j) * 512 + k0 + kc;
        float4 w = *(const float4*)gp;
        uint2 pk; pk.x = pk2(w.x, w.y); pk.y = pk2(w.z, w.w);
        *(uint2*)&sB[j * ASTRIDE + kc] = pk;
        if (k0 + 32 < 512) __builtin_prefetch(gp + 32, 0, 1);
      }
    } else {
      { // sA[i][k] = At[c][(k0+k)*512 + bi+i] : contiguous i, LDS transpose
        int k = tid >> 3, ib = (tid & 7) * 8;
        const float* gp = A + cbase + (size_t)(k0 + k) * 512 + bi + ib;
        float4 v0 = *(const float4*)gp;
        float4 v1 = *(const float4*)(gp + 4);
        sA[(ib + 0) * ASTRIDE + k] = f2bf(v0.x); sA[(ib + 1) * ASTRIDE + k] = f2bf(v0.y);
        sA[(ib + 2) * ASTRIDE + k] = f2bf(v0.z); sA[(ib + 3) * ASTRIDE + k] = f2bf(v0.w);
        sA[(ib + 4) * ASTRIDE + k] = f2bf(v1.x); sA[(ib + 5) * ASTRIDE + k] = f2bf(v1.y);
        sA[(ib + 6) * ASTRIDE + k] = f2bf(v1.z); sA[(ib + 7) * ASTRIDE + k] = f2bf(v1.w);
        if (k0 + 32 < 512) __builtin_prefetch(A + cbase + (size_t)(k0 + 32 + k) * 512 + bi + ib, 0, 1);
      }
      { // sB[j][k] = Bt[c][(k0+k)*512 + bj+j] : contiguous j, LDS transpose
        int k = tid >> 3, jb = (tid & 7) * 4;
        const float* gp = Bm + cbase + (size_t)(k0 + k) * 512 + bj + jb;
        float4 w = *(const float4*)gp;
        sB[(jb + 0) * ASTRIDE + k] = f2bf(w.x); sB[(jb + 1) * ASTRIDE + k] = f2bf(w.y);
        sB[(jb + 2) * ASTRIDE + k] = f2bf(w.z); sB[(jb + 3) * ASTRIDE + k] = f2bf(w.w);
      }
    }
    __syncthreads();
    v16bf af = frag_ld_a(&sA[(wm * 16) * ASTRIDE], ASTRIDE, lane);
    v16bf bf = frag_ld_b(&sB[(wn * 16) * ASTRIDE], ASTRIDE, lane);
    acc = wmma_bf16(af, bf, acc);
    __syncthreads();
  }
  int n = bj + wn * 16 + (lane & 15);
#pragma unroll
  for (int r = 0; r < 8; ++r) {
    int m = bi + wm * 16 + ((lane >> 4) << 3) + r;
    X[((size_t)m * 512 + n) * 128 + c] = acc[r];
  }
}

// ---------------------------------------------------------------------------
// Fused pair FFN: z += W2 @ relu(W1 @ z + b1) + b2, per 32-row tile.
// Hidden (32x512) lives in LDS as bf16; both GEMMs via WMMA.
// ---------------------------------------------------------------------------
__global__ void __launch_bounds__(256)
k_pair_ffn(float* __restrict__ z, const float* __restrict__ W1, const float* __restrict__ b1,
           const float* __restrict__ W2, const float* __restrict__ b2)
{
  __shared__ __align__(16) unsigned short sA[32 * 136];
  __shared__ __align__(16) unsigned short sW[128 * ASTRIDE];
  __shared__ __align__(16) unsigned short sH[32 * 520];
  const int tid = threadIdx.x, lane = tid & 31, wave = tid >> 5;
  const size_t row0 = (size_t)blockIdx.x * 32;
  { // stage A (z rows) as bf16: 16 contiguous elements/thread -> 2x ds_store_b128
    int r = tid >> 3, cb = (tid & 7) * 16;
    const float* gp = z + (row0 + r) * 128 + cb;
    float4 v0 = *(const float4*)(gp + 0), v1 = *(const float4*)(gp + 4);
    float4 v2 = *(const float4*)(gp + 8), v3 = *(const float4*)(gp + 12);
    uint4 p0, p1;
    p0.x = pk2(v0.x, v0.y); p0.y = pk2(v0.z, v0.w); p0.z = pk2(v1.x, v1.y); p0.w = pk2(v1.z, v1.w);
    p1.x = pk2(v2.x, v2.y); p1.y = pk2(v2.z, v2.w); p1.z = pk2(v3.x, v3.y); p1.w = pk2(v3.z, v3.w);
    *(uint4*)&sA[r * 136 + cb] = p0;
    *(uint4*)&sA[r * 136 + cb + 8] = p1;
  }
  __syncthreads();
  { // GEMM1: H = relu(A @ W1 + b1), 32x512
    const int wm = wave & 1, wn = wave >> 1;
    const int kw = tid >> 3, nw = (tid & 7) * 8;
    for (int nb = 0; nb < 8; ++nb) {
      v8f acc = {};
      for (int ks = 0; ks < 128; ks += 32) {
        {
          const float* gp = W1 + (size_t)(ks + kw) * 512 + nb * 64 + nw;
          float4 v0 = *(const float4*)gp;
          float4 v1 = *(const float4*)(gp + 4);
          sW[(nw + 0) * ASTRIDE + kw] = f2bf(v0.x); sW[(nw + 1) * ASTRIDE + kw] = f2bf(v0.y);
          sW[(nw + 2) * ASTRIDE + kw] = f2bf(v0.z); sW[(nw + 3) * ASTRIDE + kw] = f2bf(v0.w);
          sW[(nw + 4) * ASTRIDE + kw] = f2bf(v1.x); sW[(nw + 5) * ASTRIDE + kw] = f2bf(v1.y);
          sW[(nw + 6) * ASTRIDE + kw] = f2bf(v1.z); sW[(nw + 7) * ASTRIDE + kw] = f2bf(v1.w);
        }
        __syncthreads();
        v16bf af = frag_ld_a(&sA[(wm * 16) * 136 + ks], 136, lane);
        v16bf bf = frag_ld_b(&sW[(wn * 16) * ASTRIDE], ASTRIDE, lane);
        acc = wmma_bf16(af, bf, acc);
        __syncthreads();
      }
      int n = nb * 64 + wn * 16 + (lane & 15);
#pragma unroll
      for (int r = 0; r < 8; ++r) {
        int m = wm * 16 + ((lane >> 4) << 3) + r;
        sH[m * 520 + n] = f2bf(fmaxf(acc[r] + b1[n], 0.f));
      }
      __syncthreads();
    }
  }
  { // GEMM2: out = H @ W2 + b2 + z, 32x128
    const int wm = wave & 1, wn2 = (wave >> 1) * 2;
    const int kw = tid >> 3, nw = (tid & 7) * 16;
    v8f acc0 = {}, acc1 = {};
    for (int ks = 0; ks < 512; ks += 32) {
      {
        const float* gp = W2 + (size_t)(ks + kw) * 128 + nw;
        float4 v0 = *(const float4*)(gp + 0), v1 = *(const float4*)(gp + 4);
        float4 v2 = *(const float4*)(gp + 8), v3 = *(const float4*)(gp + 12);
        sW[(nw + 0) * ASTRIDE + kw] = f2bf(v0.x);  sW[(nw + 1) * ASTRIDE + kw] = f2bf(v0.y);
        sW[(nw + 2) * ASTRIDE + kw] = f2bf(v0.z);  sW[(nw + 3) * ASTRIDE + kw] = f2bf(v0.w);
        sW[(nw + 4) * ASTRIDE + kw] = f2bf(v1.x);  sW[(nw + 5) * ASTRIDE + kw] = f2bf(v1.y);
        sW[(nw + 6) * ASTRIDE + kw] = f2bf(v1.z);  sW[(nw + 7) * ASTRIDE + kw] = f2bf(v1.w);
        sW[(nw + 8) * ASTRIDE + kw] = f2bf(v2.x);  sW[(nw + 9) * ASTRIDE + kw] = f2bf(v2.y);
        sW[(nw + 10) * ASTRIDE + kw] = f2bf(v2.z); sW[(nw + 11) * ASTRIDE + kw] = f2bf(v2.w);
        sW[(nw + 12) * ASTRIDE + kw] = f2bf(v3.x); sW[(nw + 13) * ASTRIDE + kw] = f2bf(v3.y);
        sW[(nw + 14) * ASTRIDE + kw] = f2bf(v3.z); sW[(nw + 15) * ASTRIDE + kw] = f2bf(v3.w);
      }
      __syncthreads();
      v16bf af  = frag_ld_a(&sH[(wm * 16) * 520 + ks], 520, lane);
      v16bf bf0 = frag_ld_b(&sW[(wn2 * 16) * ASTRIDE], ASTRIDE, lane);
      v16bf bf1 = frag_ld_b(&sW[((wn2 + 1) * 16) * ASTRIDE], ASTRIDE, lane);
      acc0 = wmma_bf16(af, bf0, acc0);
      acc1 = wmma_bf16(af, bf1, acc1);
      __syncthreads();
    }
#pragma unroll
    for (int half = 0; half < 2; ++half) {
      v8f acc = half ? acc1 : acc0;
      int n = (wn2 + half) * 16 + (lane & 15);
#pragma unroll
      for (int r = 0; r < 8; ++r) {
        int m = wm * 16 + ((lane >> 4) << 3) + r;
        size_t idx = (row0 + m) * 128 + n;
        z[idx] = acc[r] + b2[n] + z[idx];
      }
    }
  }
}

// --------------------------- VALU helper kernels ---------------------------

__global__ void __launch_bounds__(128)
k_ln(const float* __restrict__ in, float* __restrict__ out,
     const float* __restrict__ g, const float* __restrict__ b, int D)
{
  __shared__ float r1[128], r2[128];
  int tid = threadIdx.x;
  const float* row = in + (size_t)blockIdx.x * D;
  float s = 0.f, s2 = 0.f;
  for (int c = tid; c < D; c += 128) { float v = row[c]; s += v; s2 += v * v; }
  r1[tid] = s; r2[tid] = s2; __syncthreads();
  for (int st = 64; st > 0; st >>= 1) {
    if (tid < st) { r1[tid] += r1[tid + st]; r2[tid] += r2[tid + st]; }
    __syncthreads();
  }
  float mean = r1[0] / D;
  float var = r2[0] / D - mean * mean;
  float inv = rsqrtf(var + 1e-5f);
  float* orow = out + (size_t)blockIdx.x * D;
  for (int c = tid; c < D; c += 128) orow[c] = (row[c] - mean) * inv * g[c] + b[c];
}

// rigid transform of point-linear outputs
__global__ void k_pts(const float* __restrict__ lin, int np,
                      const float* __restrict__ rots, const float* __restrict__ trans,
                      float* __restrict__ koutA, float* __restrict__ koutB,
                      int psplit, int total)
{
  int idx = blockIdx.x * blockDim.x + threadIdx.x;
  if (idx >= total) return;
  int p = idx % np; int h = (idx / np) % 12; int n = idx / (np * 12);
  int cols = 12 * np;
  const float* L = lin + (size_t)n * (3 * cols) + h * np + p;
  float t0 = L[0], t1 = L[cols], t2 = L[2 * cols];
  const float* R = rots + n * 9;
  float o0 = R[0]*t0 + R[1]*t1 + R[2]*t2 + trans[n*3+0];
  float o1 = R[3]*t0 + R[4]*t1 + R[5]*t2 + trans[n*3+1];
  float o2 = R[6]*t0 + R[7]*t1 + R[8]*t2 + trans[n*3+2];
  float* dst; int pp, npd;
  if (p < psplit) { dst = koutA; pp = p; npd = psplit; }
  else            { dst = koutB; pp = p - psplit; npd = np - psplit; }
  size_t o = (((size_t)n * 12 + h) * npd + pp) * 3;
  dst[o] = o0; dst[o + 1] = o1; dst[o + 2] = o2;
}

// batt[(i*512+j)*12+h] = z[i,j,:] . Wb[:,h] + bb[h]
__global__ void __launch_bounds__(256)
k_bias_att(const float* __restrict__ z, const float* __restrict__ Wb,
           const float* __restrict__ bb, float* __restrict__ batt)
{
  __shared__ float sz[16 * 128];
  __shared__ float sw[128 * 12];
  int tid = threadIdx.x;
  size_t row0 = (size_t)blockIdx.x * 16;
  { // 2048 floats, 8 contiguous per thread
    int idx = tid * 8;
    float4 a = *(const float4*)&z[row0 * 128 + idx];
    float4 b = *(const float4*)&z[row0 * 128 + idx + 4];
    *(float4*)&sz[idx] = a; *(float4*)&sz[idx + 4] = b;
  }
  if (tid < 192) { // 1536 floats
    int idx = tid * 8;
    float4 a = *(const float4*)&Wb[idx];
    float4 b = *(const float4*)&Wb[idx + 4];
    *(float4*)&sw[idx] = a; *(float4*)&sw[idx + 4] = b;
  }
  __syncthreads();
  if (tid < 192) {
    int r = tid / 12, h = tid % 12;
    float acc = bb[h];
    for (int k = 0; k < 128; ++k) acc += sz[r * 128 + k] * sw[k * 12 + h];
    batt[(row0 + r) * 12 + h] = acc;
  }
}

__global__ void __launch_bounds__(256)
k_att_logits(const float* __restrict__ LINQ, const float* __restrict__ LINKV,
             const float* __restrict__ QP, const float* __restrict__ KP,
             const float* __restrict__ batt, const float* __restrict__ mask,
             const float* __restrict__ hwraw, float* __restrict__ att)
{
  __shared__ float qs[32][36], ks_[32][36];
  __shared__ float qp[32][12], kp[32][12];
  __shared__ float qn[32], kn[32];
  int tid = threadIdx.x;
  int bi = blockIdx.x * 32, bj = blockIdx.y * 32, h = blockIdx.z;
  { // one float4 per thread per array
    int r = tid >> 3, cb = (tid & 7) * 4;
    *(float4*)&qs[r][cb]  = *(const float4*)&LINQ[(size_t)(bi + r) * 384 + h * 32 + cb];
    *(float4*)&ks_[r][cb] = *(const float4*)&LINKV[(size_t)(bj + r) * 768 + h * 64 + cb];
  }
  for (int idx = tid; idx < 384; idx += 256) {
    int r = idx / 12, mm = idx % 12;
    qp[r][mm] = QP[((size_t)(bi + r) * 12 + h) * 12 + mm];
    kp[r][mm] = KP[((size_t)(bj + r) * 12 + h) * 12 + mm];
  }
  __syncthreads();
  if (tid < 64) {
    int r = tid & 31;
    float s = 0.f;
    if (tid < 32) { for (int mm = 0; mm < 12; ++mm) s += qp[r][mm] * qp[r][mm]; qn[r] = s; }
    else          { for (int mm = 0; mm < 12; ++mm) s += kp[r][mm] * kp[r][mm]; kn[r] = s; }
  }
  __syncthreads();
  float hw = log1pf(__expf(hwraw[h])) * 0.13608276348795434f; // sqrt(1/54)
  const float sc_qk = 0.10206207261596575f;                   // sqrt(1/96)
  const float sc_b  = 0.57735026918962576f;                   // sqrt(1/3)
#pragma unroll
  for (int it = 0; it < 4; ++it) {
    int idx = tid + it * 256;
    int ii = idx >> 5, jj = idx & 31;
    float dot = 0.f;
    for (int cc = 0; cc < 32; ++cc) dot += qs[ii][cc] * ks_[jj][cc];
    float pdot = 0.f;
    for (int mm = 0; mm < 12; ++mm) pdot += qp[ii][mm] * kp[jj][mm];
    float pt = qn[ii] + kn[jj] - 2.f * pdot;
    float mterm = (mask[bi + ii] * mask[bj + jj] - 1.f) * 100000.f;
    size_t pr = (size_t)(bi + ii) * 512 + (bj + jj);
    att[(size_t)h * 262144 + pr] = dot * sc_qk + sc_b * batt[pr * 12 + h] - 0.5f * hw * pt + mterm;
  }
}

__global__ void __launch_bounds__(256)
k_softmax512(float* __restrict__ att)
{
  __shared__ float red[256];
  int tid = threadIdx.x;
  float* row = att + (size_t)blockIdx.x * 512;
  float a = row[tid], b = row[tid + 256];
  red[tid] = fmaxf(a, b); __syncthreads();
  for (int s = 128; s > 0; s >>= 1) { if (tid < s) red[tid] = fmaxf(red[tid], red[tid + s]); __syncthreads(); }
  float mx = red[0]; __syncthreads();
  a = __expf(a - mx); b = __expf(b - mx);
  red[tid] = a + b; __syncthreads();
  for (int s = 128; s > 0; s >>= 1) { if (tid < s) red[tid] += red[tid + s]; __syncthreads(); }
  float inv = 1.f / red[0];
  row[tid] = a * inv; row[tid + 256] = b * inv;
}

// o and raw o_pt accumulation for one (i,h)
__global__ void __launch_bounds__(256)
k_attn_out(const float* __restrict__ att, const float* __restrict__ LINKV,
           const float* __restrict__ VP, float* __restrict__ feats,
           float* __restrict__ optraw)
{
  __shared__ float sa[512];
  __shared__ float psum[224];
  int tid = threadIdx.x;
  int i = blockIdx.x, h = blockIdx.y;
  if (tid < 128)
    *(float4*)&sa[tid * 4] = *(const float4*)&att[(size_t)h * 262144 + (size_t)i * 512 + tid * 4];
  __syncthreads();
  if (tid < 224) {
    int oi = tid % 56, jq = tid / 56;
    float acc = 0.f;
    for (int j = jq * 128; j < jq * 128 + 128; ++j) {
      float v = (oi < 32) ? LINKV[(size_t)j * 768 + h * 64 + 32 + oi]
                          : VP[((size_t)j * 12 + h) * 24 + (oi - 32)];
      acc += sa[j] * v;
    }
    psum[tid] = acc;
  }
  __syncthreads();
  if (tid < 56) {
    float s = psum[tid] + psum[tid + 56] + psum[tid + 112] + psum[tid + 168];
    if (tid < 32) feats[(size_t)i * 2304 + h * 32 + tid] = s;
    else          optraw[((size_t)i * 12 + h) * 24 + (tid - 32)] = s;
  }
}

__global__ void k_opt_finish(const float* __restrict__ optraw, const float* __restrict__ rots,
                             const float* __restrict__ trans, float* __restrict__ feats)
{
  int idx = blockIdx.x * blockDim.x + threadIdx.x;
  if (idx >= 512 * 96) return;
  int hp = idx % 96, n = idx / 96;
  int h = hp / 8, p = hp % 8;
  const float* raw = optraw + ((size_t)n * 12 + h) * 24 + p * 3;
  float v0 = raw[0] - trans[n * 3 + 0];
  float v1 = raw[1] - trans[n * 3 + 1];
  float v2 = raw[2] - trans[n * 3 + 2];
  const float* R = rots + n * 9;
  float o0 = R[0] * v0 + R[3] * v1 + R[6] * v2; // R^T v
  float o1 = R[1] * v0 + R[4] * v1 + R[7] * v2;
  float o2 = R[2] * v0 + R[5] * v1 + R[8] * v2;
  int hp8 = h * 8 + p;
  size_t base = (size_t)n * 2304;
  feats[base + 384 + hp8] = o0;
  feats[base + 480 + hp8] = o1;
  feats[base + 576 + hp8] = o2;
  feats[base + 672 + hp8] = sqrtf(o0 * o0 + o1 * o1 + o2 * o2 + 1e-8f);
}

// o_pair[i,h,zc] = sum_j a[h,i,j] * z[i,j,zc]
__global__ void __launch_bounds__(256)
k_opair(const float* __restrict__ att, const float* __restrict__ z, float* __restrict__ feats)
{
  __shared__ float sa[12 * 512];
  int tid = threadIdx.x;
  int i = blockIdx.x;
#pragma unroll
  for (int it = 0; it < 6; ++it) {
    int idx4 = (tid + it * 256) * 4;
    int h = idx4 >> 9, j = idx4 & 511;
    *(float4*)&sa[idx4] = *(const float4*)&att[(size_t)h * 262144 + (size_t)i * 512 + j];
  }
  __syncthreads();
  int zc = tid & 127, hg = tid >> 7;
  float acc[6] = {0, 0, 0, 0, 0, 0};
  for (int j = 0; j < 512; ++j) {
    float zv = z[((size_t)i * 512 + j) * 128 + zc];
#pragma unroll
    for (int e = 0; e < 6; ++e) acc[e] += sa[(hg + 2 * e) * 512 + j] * zv;
  }
  for (int e = 0; e < 6; ++e)
    feats[(size_t)i * 2304 + 768 + (hg + 2 * e) * 128 + zc] = acc[e];
}

// cat([q_i*k_j, q_i-k_j]) features for s2p; 4 contiguous channels per thread
__global__ void k_catfeat(const float* __restrict__ qk, float* __restrict__ out, long long total4)
{
  long long t = (long long)blockIdx.x * blockDim.x + threadIdx.x;
  if (t >= total4) return;
  long long idx = t * 4;
  int cc = (int)(idx & 127);
  long long r = idx >> 7;
  int i = (int)(r >> 9), j = (int)(r & 511);
  float4 o;
  if (cc < 64) {
    float4 q = *(const float4*)&qk[i * 128 + cc];
    float4 k = *(const float4*)&qk[j * 128 + 64 + cc];
    o = make_float4(q.x * k.x, q.y * k.y, q.z * k.z, q.w * k.w);
  } else {
    int c2 = cc - 64;
    float4 q = *(const float4*)&qk[i * 128 + c2];
    float4 k = *(const float4*)&qk[j * 128 + 64 + c2];
    o = make_float4(q.x - k.x, q.y - k.y, q.z - k.z, q.w - k.w);
  }
  *(float4*)&out[idx] = o;
}

__global__ void k_fu_gemv(const float* __restrict__ A, const float* __restrict__ W,
                          const float* __restrict__ b, float* __restrict__ out)
{
  int n = blockIdx.x, t = threadIdx.x;
  if (t < 6) {
    float acc = b[t];
    for (int k = 0; k < 384; ++k) acc += A[(size_t)n * 384 + k] * W[k * 6 + t];
    out[n * 6 + t] = acc;
  }
}

__global__ void k_frame(const float* __restrict__ U, const float* __restrict__ rots,
                        const float* __restrict__ trans, float* __restrict__ rots_out,
                        float* __restrict__ trans_out)
{
  int n = blockIdx.x * blockDim.x + threadIdx.x;
  if (n >= 512) return;
  const float* u = U + n * 6;
  float vx = u[0], vy = u[1], vz = u[2], rx = u[3], ry = u[4], rz = u[5];
  const float* R0 = rots + n * 9;
  trans_out[n*3+0] = trans[n*3+0] + R0[0]*vx + R0[1]*vy + R0[2]*vz;
  trans_out[n*3+1] = trans[n*3+1] + R0[3]*vx + R0[4]*vy + R0[5]*vz;
  trans_out[n*3+2] = trans[n*3+2] + R0[6]*vx + R0[7]*vy + R0[8]*vz;
  float t2 = rx*rx + ry*ry + rz*rz;
  float th = sqrtf(t2 + 1e-12f);
  float a = sinf(th) / th;
  float cf = (1.f - cosf(th)) / (th * th);
  float K[9] = {0.f, -rz, ry,  rz, 0.f, -rx,  -ry, rx, 0.f};
  float K2[9];
  for (int i_ = 0; i_ < 3; ++i_)
    for (int j_ = 0; j_ < 3; ++j_) {
      float s = 0.f;
      for (int k_ = 0; k_ < 3; ++k_) s += K[i_*3+k_] * K[k_*3+j_];
      K2[i_*3+j_] = s;
    }
  float R[9];
  for (int i_ = 0; i_ < 3; ++i_)
    for (int j_ = 0; j_ < 3; ++j_)
      R[i_*3+j_] = (i_ == j_ ? 1.f : 0.f) + a * K[i_*3+j_] + cf * K2[i_*3+j_];
  for (int i_ = 0; i_ < 3; ++i_)
    for (int k_ = 0; k_ < 3; ++k_) {
      float s = 0.f;
      for (int j_ = 0; j_ < 3; ++j_) s += R0[i_*3+j_] * R[k_*3+j_];
      rots_out[n*9 + i_*3 + k_] = s;
    }
}

// ---------------------------------------------------------------------------

extern "C" void kernel_launch(void* const* d_in, const int* in_sizes, int n_in,
                              void* d_out, int out_size, void* d_ws, size_t ws_size,
                              hipStream_t stream) {
  (void)in_sizes; (void)n_in; (void)out_size; (void)ws_size;
  auto F = [&](int i) { return (const float*)d_in[i]; };
  const float* x     = F(0);
  const float* zin   = F(1);
  const float* mask  = F(2);
  const float* rots  = F(3);
  const float* trans = F(4);

  float* outf      = (float*)d_out;
  float* out_x     = outf;
  float* out_z     = outf + 196608;
  float* out_mask  = out_z + 33554432;
  float* out_rots  = out_mask + 512;
  float* out_trans = out_rots + 4608;

  char* wsb = (char*)d_ws;
  size_t off = 0;
  auto alloc = [&](size_t nElem) {
    float* p = (float*)(wsb + off);
    off += ((nElem * sizeof(float) + 255) & ~(size_t)255);
    return p;
  };
  float* S1     = alloc(196608);
  float* LINQ   = alloc(196608);
  float* LINKV  = alloc(393216);
  float* QPTL   = alloc(73728);
  float* KVPTL  = alloc(221184);
  float* QP     = alloc(73728);
  float* KP     = alloc(73728);
  float* VP     = alloc(147456);
  float* BATT   = alloc(3145728);
  float* ATT    = alloc(3145728);
  float* FEATS  = alloc(1179648);
  float* OPTRAW = alloc(147456);
  float* X1     = alloc(196608);
  float* H1     = alloc(196608);
  float* FFH    = alloc(786432);
  float* X2     = alloc(196608);
  float* XL     = alloc(196608);
  float* QK     = alloc(65536);
  float* U      = alloc(3072);
  float* B1     = alloc(33554432);
  float* B2     = alloc(33554432);
  float* B3     = alloc(33554432);
  float* B4     = alloc(33554432);
  float* B5     = alloc(33554432);

  auto gemm = [&](int epi, bool tc, const float* A, int lda, const float* W, int ldw,
                  const float* bias, float* C, int ldc, int M, int N, int K,
                  const float* mul, const float* res, const float* mv, int mN) {
    dim3 grid((N + 63) / 64, (M + 63) / 64);
    if (epi == 3 && tc) {
      k_gemm<3, true><<<grid, 256, 0, stream>>>(A, lda, W, ldw, bias, C, ldc, M, N, K, mul, res, mv, mN);
      return;
    }
    switch (epi) {
      case 0: k_gemm<0, false><<<grid, 256, 0, stream>>>(A, lda, W, ldw, bias, C, ldc, M, N, K, mul, res, mv, mN); break;
      case 1: k_gemm<1, false><<<grid, 256, 0, stream>>>(A, lda, W, ldw, bias, C, ldc, M, N, K, mul, res, mv, mN); break;
      case 2: k_gemm<2, false><<<grid, 256, 0, stream>>>(A, lda, W, ldw, bias, C, ldc, M, N, K, mul, res, mv, mN); break;
      case 3: k_gemm<3, false><<<grid, 256, 0, stream>>>(A, lda, W, ldw, bias, C, ldc, M, N, K, mul, res, mv, mN); break;
      case 4: k_gemm<4, false><<<grid, 256, 0, stream>>>(A, lda, W, ldw, bias, C, ldc, M, N, K, mul, res, mv, mN); break;
      case 5: k_gemm<5, false><<<grid, 256, 0, stream>>>(A, lda, W, ldw, bias, C, ldc, M, N, K, mul, res, mv, mN); break;
    }
  };

  // ---------------- IPA ----------------
  k_ln<<<512, 128, 0, stream>>>(x, S1, F(5), F(6), 384);
  gemm(0, false, S1, 384, F(7),  384, F(8),  LINQ,  384, 512, 384, 384, nullptr, nullptr, nullptr, 0);
  gemm(0, false, S1, 384, F(9),  768, F(10), LINKV, 768, 512, 768, 384, nullptr, nullptr, nullptr, 0);
  gemm(0, false, S1, 384, F(11), 144, F(12), QPTL,  144, 512, 144, 384, nullptr, nullptr, nullptr, 0);
  gemm(0, false, S1, 384, F(13), 432, F(14), KVPTL, 432, 512, 432, 384, nullptr, nullptr, nullptr, 0);
  k_pts<<<(512 * 12 * 4 + 255) / 256, 256, 0, stream>>>(QPTL, 4, rots, trans, QP, nullptr, 4, 512 * 12 * 4);
  k_pts<<<(512 * 12 * 12 + 255) / 256, 256, 0, stream>>>(KVPTL, 12, rots, trans, KP, VP, 4, 512 * 12 * 12);
  k_bias_att<<<16384, 256, 0, stream>>>(zin, F(15), F(16), BATT);
  k_att_logits<<<dim3(16, 16, 12), 256, 0, stream>>>(LINQ, LINKV, QP, KP, BATT, mask, F(17), ATT);
  k_softmax512<<<6144, 256, 0, stream>>>(ATT);
  k_attn_out<<<dim3(512, 12), 256, 0, stream>>>(ATT, LINKV, VP, FEATS, OPTRAW);
  k_opt_finish<<<192, 256, 0, stream>>>(OPTRAW, rots, trans, FEATS);
  k_opair<<<512, 256, 0, stream>>>(ATT, zin, FEATS);
  gemm(5, false, FEATS, 2304, F(18), 384, F(19), X1, 384, 512, 384, 2304, nullptr, x, nullptr, 0);

  // ---------------- FF ----------------
  k_ln<<<512, 128, 0, stream>>>(X1, H1, F(20), F(21), 384);
  gemm(1, false, H1, 384, F(22), 1536, F(23), FFH, 1536, 512, 1536, 384, nullptr, nullptr, nullptr, 0);
  gemm(5, false, FFH, 1536, F(24), 384, F(25), X2, 384, 512, 384, 1536, nullptr, X1, nullptr, 0);

  // ---------------- s2p ----------------
  k_ln<<<512, 128, 0, stream>>>(X2, XL, F(26), F(27), 384);
  gemm(0, false, XL, 384, F(28), 128, F(29), QK, 128, 512, 128, 384, nullptr, nullptr, nullptr, 0);
  k_catfeat<<<32768, 256, 0, stream>>>(QK, B1, (long long)262144 * 128 / 4);
  gemm(5, false, B1, 128, F(30), 128, F(31), out_z, 128, 262144, 128, 128, nullptr, zin, nullptr, 0);

  // ---------------- triangle updates ----------------
  auto tri = [&](int t, bool outg) {
    k_ln<<<262144, 128, 0, stream>>>(out_z, B1, F(t + 0), F(t + 1), 128);
    gemm(0, false, B1, 128, F(t + 4), 128, F(t + 5), B4, 128, 262144, 128, 128, nullptr, nullptr, nullptr, 0);
    // gate tensors written channel-major for coalesced triangle-GEMM staging
    gemm(3, true,  B1, 128, F(t + 2), 128, F(t + 3), B2, 128, 262144, 128, 128, B4, nullptr, mask, 512);
    gemm(0, false, B1, 128, F(t + 8), 128, F(t + 9), B4, 128, 262144, 128, 128, nullptr, nullptr, nullptr, 0);
    gemm(3, true,  B1, 128, F(t + 6), 128, F(t + 7), B3, 128, 262144, 128, 128, B4, nullptr, mask, 512);
    gemm(2, false, B1, 128, F(t + 10), 128, F(t + 11), B5, 128, 262144, 128, 128, nullptr, nullptr, nullptr, 0);
    if (outg) k_tri_mm<true><<<dim3(128, 16, 8), 256, 0, stream>>>(B2, B3, B4);
    else      k_tri_mm<false><<<dim3(128, 16, 8), 256, 0, stream>>>(B2, B3, B4);
    k_ln<<<262144, 128, 0, stream>>>(B4, B1, F(t + 12), F(t + 13), 128);
    gemm(4, false, B1, 128, F(t + 14), 128, F(t + 15), out_z, 128, 262144, 128, 128, B5, out_z, nullptr, 0);
  };
  tri(32, true);
  tri(48, false);

  // ---------------- pair FFN ----------------
  k_pair_ffn<<<8192, 256, 0, stream>>>(out_z, F(64), F(65), F(66), F(67));

  // ---------------- frame update + outputs ----------------
  k_ln<<<512, 128, 0, stream>>>(X2, XL, F(68), F(69), 384);
  k_fu_gemv<<<512, 32, 0, stream>>>(XL, F(70), F(71), U);
  k_frame<<<2, 256, 0, stream>>>(U, rots, trans, out_rots, out_trans);
  hipMemcpyAsync(out_x, X2, 196608 * sizeof(float), hipMemcpyDeviceToDevice, stream);
  hipMemcpyAsync(out_mask, mask, 512 * sizeof(float), hipMemcpyDeviceToDevice, stream);
}
